// BahdanauAttention_13615046328779
// MI455X (gfx1250) — matrix operand order repaired and checked by reading, not compiled, not run on hardware
//
#include <hip/hip_runtime.h>
#include <hip/hip_bf16.h>
#include <math.h>

#define B_    32
#define S_    2048
#define ENC_  1024
#define DEC_  1024
#define ATTN_ 512
#define STILE 128

typedef __attribute__((ext_vector_type(16))) __bf16 v16bf;
typedef __attribute__((ext_vector_type(8)))  float  v8f;

union Frag { uint4 u[2]; v16bf v; };

// round-to-nearest-even f32 -> bf16, packed pair into one dword
__device__ __forceinline__ unsigned int pk_bf16(float a, float b) {
  unsigned int ua = __float_as_uint(a);
  unsigned int ub = __float_as_uint(b);
  ua += 0x7FFFu + ((ua >> 16) & 1u);
  ub += 0x7FFFu + ((ub >> 16) & 1u);
  return (ua >> 16) | (ub & 0xFFFF0000u);
}

// branch-free tanh: 1 - 2*rcp(e^{2x}+1).  e->inf => 1; e->0 => -1; NaN prop.
// v_exp_f32 + v_rcp_f32 are TRANS ops that co-execute with WMMA.
__device__ __forceinline__ float fast_tanh(float x) {
  float e = __expf(2.0f * x);
  return 1.0f - 2.0f * __builtin_amdgcn_rcpf(e + 1.0f);
}

// ---------------------------------------------------------------------------
// Kernel P: pre-swizzle W_h (f32 [ATTN][ENC]) into bf16 B-fragment layout.
// wsW uint4 piece g: q=g&1, lane=(g>>1)&31, ks=(g>>6)&31, tile t=g>>11.
// Piece = 8 bf16 of row (16t + lane&15), cols 32ks + 16*(lane>>4) + 8q.
// ---------------------------------------------------------------------------
__global__ void pack_w_kernel(const float* __restrict__ Wh,
                              uint4* __restrict__ wsW) {
  const int g  = blockIdx.x * 256 + threadIdx.x;   // 65536 pieces
  const int q  = g & 1;
  const int l  = (g >> 1) & 31;
  const int ks = (g >> 6) & 31;
  const int t  = g >> 11;
  const int row = t * 16 + (l & 15);
  const int col = ks * 32 + 16 * (l >> 4) + 8 * q;
  const float4* src = (const float4*)(Wh + (size_t)row * ENC_ + col);
  float4 f0 = src[0], f1 = src[1];
  uint4 o;
  o.x = pk_bf16(f0.x, f0.y); o.y = pk_bf16(f0.z, f0.w);
  o.z = pk_bf16(f1.x, f1.y); o.w = pk_bf16(f1.z, f1.w);
  wsW[g] = o;
}

// ---------------------------------------------------------------------------
// Kernel A: proj_s[b,a] = sum_d W_s[a,d] * dec[b,d]     (tiny, VALU)
// ---------------------------------------------------------------------------
__global__ void proj_s_kernel(const float* __restrict__ dec,
                              const float* __restrict__ Ws,
                              float* __restrict__ ps) {
  const int b = blockIdx.x, tid = threadIdx.x;
  __shared__ float h[DEC_];
  for (int i = tid; i < DEC_; i += 256) h[i] = dec[b * DEC_ + i];
  __syncthreads();
  for (int a = tid; a < ATTN_; a += 256) {
    const float4* wr = (const float4*)(Ws + (size_t)a * DEC_);
    float acc = 0.f;
    #pragma unroll 4
    for (int d = 0; d < DEC_ / 4; ++d) {
      float4 w = wr[d];
      acc += w.x * h[4*d] + w.y * h[4*d+1] + w.z * h[4*d+2] + w.w * h[4*d+3];
    }
    ps[b * ATTN_ + a] = acc;
  }
}

// ---------------------------------------------------------------------------
// Kernel B: fused  score[b,s] = v . tanh(W_h @ enc[b,s] + ps[b])
// 256 threads = 8 waves; wave w owns s-rows [sBase+16w, sBase+16w+16).
// enc tile (128x1024) staged once as bf16 in A-fragment-packed LDS
// (lane-consecutive ds_load_b128, conflict-free).  Pre-swizzled bf16 W tiles
// are register-double-buffered.  Per-lane score partials accumulate across
// all a-tiles; single cross-lane shuffle reduction at the end.
// ---------------------------------------------------------------------------
__global__ __launch_bounds__(256, 1)
void score_kernel(const float* __restrict__ enc, const uint4* __restrict__ wsW,
                  const float* __restrict__ ps,  const float* __restrict__ vv,
                  float* __restrict__ scores) {
  __shared__ __align__(16) uint4 ldsEnc[STILE * ENC_ * 2 / 16]; // 256KB
  __shared__ __align__(16) uint4 ldsW[2048];                    // 32KB
  const int b     = blockIdx.y;
  const int sBase = blockIdx.x * STILE;
  const int tid   = threadIdx.x;

  // ---- stage encoder tile f32 -> bf16, A-fragment-packed ----
  const float* encB = enc + (size_t)b * S_ * ENC_ + (size_t)sBase * ENC_;
  #pragma unroll 2
  for (int p = tid; p < STILE * ENC_ * 2 / 16; p += 256) {
    const int q  = p & 1;
    const int l  = (p >> 1) & 31;
    const int ks = (p >> 6) & 31;
    const int rg = p >> 11;
    const int row = rg * 16 + (l & 15);
    const int col = ks * 32 + 16 * (l >> 4) + 8 * q;
    const float4* src = (const float4*)(encB + (size_t)row * ENC_ + col);
    float4 f0 = src[0], f1 = src[1];
    uint4 o;
    o.x = pk_bf16(f0.x, f0.y); o.y = pk_bf16(f0.z, f0.w);
    o.z = pk_bf16(f1.x, f1.y); o.w = pk_bf16(f1.z, f1.w);
    ldsEnc[p] = o;
  }

  // prefetch first W tile into registers
  uint4 wreg[8];
  #pragma unroll
  for (int i = 0; i < 8; ++i) wreg[i] = wsW[tid + i * 256];

  const int wave = tid >> 5;
  const int lane = tid & 31;
  const int half = lane >> 4;
  const int ln   = lane & 15;

  float scoreAcc[8];
  #pragma unroll
  for (int r = 0; r < 8; ++r) scoreAcc[r] = 0.f;

  const float* psB = ps + b * ATTN_;
  const uint4* aP  = ldsEnc + wave * 2048 + lane * 2;  // + 64 per k-step
  const uint4* bP  = ldsW + lane * 2;                  // + 64 per k-step

  for (int t = 0; t < ATTN_ / 16; ++t) {
    __syncthreads();                       // all waves done with previous ldsW
    #pragma unroll
    for (int i = 0; i < 8; ++i) ldsW[tid + i * 256] = wreg[i];
    __syncthreads();
    if (t + 1 < ATTN_ / 16) {              // overlap next W-tile load w/ k-loop
      #pragma unroll
      for (int i = 0; i < 8; ++i) wreg[i] = wsW[(t + 1) * 2048 + tid + i * 256];
    }
    if (t + 2 < ATTN_ / 16)                // L2 prefetch two tiles ahead
      __builtin_prefetch(wsW + (t + 2) * 2048 + tid * 8, 0, 0);

    v8f acc0 = {0.f,0.f,0.f,0.f,0.f,0.f,0.f,0.f};
    v8f acc1 = {0.f,0.f,0.f,0.f,0.f,0.f,0.f,0.f};
    #pragma unroll 2
    for (int ks = 0; ks < 32; ks += 2) {   // two independent K chains
      Frag af0, bf0, af1, bf1;
      af0.u[0] = aP[ks * 64];      af0.u[1] = aP[ks * 64 + 1];
      bf0.u[0] = bP[ks * 64];      bf0.u[1] = bP[ks * 64 + 1];
      af1.u[0] = aP[ks * 64 + 64]; af1.u[1] = aP[ks * 64 + 65];
      bf1.u[0] = bP[ks * 64 + 64]; bf1.u[1] = bP[ks * 64 + 65];
      acc0 = __builtin_amdgcn_wmma_f32_16x16x32_bf16(
          false, af0.v, false, bf0.v, (short)0, acc0, false, false);
      acc1 = __builtin_amdgcn_wmma_f32_16x16x32_bf16(
          false, af1.v, false, bf1.v, (short)0, acc1, false, false);
    }

    // C layout: acc[r] = (m = r + 8*half, n = ln); keep per-lane partials,
    // defer the cross-lane sum over n to a single reduction after the loop.
    const float psn = psB[t * 16 + ln];
    const float vn  = vv[t * 16 + ln];
    #pragma unroll
    for (int r = 0; r < 8; ++r)
      scoreAcc[r] += vn * fast_tanh(acc0[r] + acc1[r] + psn);
  }

  // one shuffle-tree reduction over the 16 attention columns per half
  #pragma unroll
  for (int r = 0; r < 8; ++r) {
    float v0 = scoreAcc[r];
    v0 += __shfl_xor(v0, 1, 32);
    v0 += __shfl_xor(v0, 2, 32);
    v0 += __shfl_xor(v0, 4, 32);
    v0 += __shfl_xor(v0, 8, 32);
    scoreAcc[r] = v0;
  }
  if (ln == 0) {
    #pragma unroll
    for (int r = 0; r < 8; ++r)
      scores[(size_t)b * S_ + sBase + wave * 16 + r + 8 * half] = scoreAcc[r];
  }
}

// ---------------------------------------------------------------------------
// Kernel C: masked softmax over S per batch
// ---------------------------------------------------------------------------
__global__ void softmax_kernel(const float* __restrict__ scores,
                               const unsigned char* __restrict__ mask,
                               float* __restrict__ attn) {
  const int b = blockIdx.x, tid = threadIdx.x;
  __shared__ float red[8];
  float m = -INFINITY;
  for (int s = tid; s < S_; s += 256) {
    float sc = mask[b * S_ + s] ? scores[b * S_ + s] : -INFINITY;
    m = fmaxf(m, sc);
  }
  #pragma unroll
  for (int o = 16; o; o >>= 1) m = fmaxf(m, __shfl_xor(m, o, 32));
  if ((tid & 31) == 0) red[tid >> 5] = m;
  __syncthreads();
  float bm = red[0];
  #pragma unroll
  for (int i = 1; i < 8; ++i) bm = fmaxf(bm, red[i]);
  __syncthreads();

  float sum = 0.f;
  for (int s = tid; s < S_; s += 256) {
    float sc = mask[b * S_ + s] ? scores[b * S_ + s] : -INFINITY;
    sum += expf(sc - bm);
  }
  #pragma unroll
  for (int o = 16; o; o >>= 1) sum += __shfl_xor(sum, o, 32);
  if ((tid & 31) == 0) red[tid >> 5] = sum;
  __syncthreads();
  float total = 0.f;
  #pragma unroll
  for (int i = 0; i < 8; ++i) total += red[i];
  const float inv = 1.0f / total;

  for (int s = tid; s < S_; s += 256) {
    float sc = mask[b * S_ + s] ? scores[b * S_ + s] : -INFINITY;
    attn[b * S_ + s] = expf(sc - bm) * inv;
  }
}

// ---------------------------------------------------------------------------
// Kernel D: context[b,e] = sum_s attn[b,s] * enc[b,s,e]   (2nd enc stream)
// ---------------------------------------------------------------------------
__global__ void context_kernel(const float* __restrict__ enc,
                               const float* __restrict__ attn,
                               float* __restrict__ ctx) {
  const int b = blockIdx.y;
  const int e = blockIdx.x * 256 + threadIdx.x;
  __shared__ float w[S_];
  for (int s = threadIdx.x; s < S_; s += 256) w[s] = attn[b * S_ + s];
  __syncthreads();
  const float* encB = enc + (size_t)b * S_ * ENC_ + e;
  float acc = 0.f;
  #pragma unroll 4
  for (int s = 0; s < S_; ++s) acc = fmaf(w[s], encB[(size_t)s * ENC_], acc);
  ctx[b * ENC_ + e] = acc;
}

// ---------------------------------------------------------------------------
extern "C" void kernel_launch(void* const* d_in, const int* in_sizes, int n_in,
                              void* d_out, int out_size, void* d_ws, size_t ws_size,
                              hipStream_t stream) {
  const float*         enc  = (const float*)d_in[0];         // [B,S,ENC]
  const float*         dec  = (const float*)d_in[1];         // [B,DEC]
  const unsigned char* mask = (const unsigned char*)d_in[2]; // [B,S] bool
  const float*         Wh   = (const float*)d_in[3];         // [ATTN,ENC]
  const float*         Ws   = (const float*)d_in[4];         // [ATTN,DEC]
  const float*         v    = (const float*)d_in[5];         // [1,ATTN]

  float* ctx  = (float*)d_out;                 // [B,ENC]
  float* attn = (float*)d_out + B_ * ENC_;     // [B,S]

  float* scores = (float*)d_ws;                // [B,S]       (256 KB)
  float* ps     = scores + B_ * S_;            // [B,ATTN]    (64 KB)
  uint4* wsW    = (uint4*)(ps + B_ * ATTN_);   // packed bf16 W_h (1 MB)

  pack_w_kernel <<<dim3(256),              256, 0, stream>>>(Wh, wsW);
  proj_s_kernel <<<dim3(B_),               256, 0, stream>>>(dec, Ws, ps);
  score_kernel  <<<dim3(S_ / STILE, B_),   256, 0, stream>>>(enc, wsW, ps, v, scores);
  softmax_kernel<<<dim3(B_),               256, 0, stream>>>(scores, mask, attn);
  context_kernel<<<dim3(ENC_ / 256, B_),   256, 0, stream>>>(enc, attn, ctx);
}
